// TransformerEncoderBlock_20066087207153
// MI455X (gfx1250) — matrix operand order repaired and checked by reading, not compiled, not run on hardware
//
#include <hip/hip_runtime.h>
#include <cstdint>

// ---------------- CDNA5 (gfx1250, wave32) types ----------------
typedef __attribute__((ext_vector_type(16))) __bf16 v16bf;
typedef __attribute__((ext_vector_type(8)))  __bf16 v8bf;
typedef __attribute__((ext_vector_type(8)))  float  v8f;

union U16 { v16bf v; v8bf h[2]; };

__device__ __forceinline__ v8f vzero() {
    v8f z = {0.f, 0.f, 0.f, 0.f, 0.f, 0.f, 0.f, 0.f};
    return z;
}

__device__ __forceinline__ v8f wmma_bf16(v16bf a, v16bf b, v8f c) {
    // D = A(16x32 bf16) * B(32x16 bf16) + C(16x16 f32)
    return __builtin_amdgcn_wmma_f32_16x16x32_bf16(
        false, a, false, b, (short)0, c, false, false);
}

// ---- CDNA5 async memory->LDS DMA (ASYNCcnt-tracked, no VGPR data path) ----
__device__ __forceinline__ unsigned ldsOff(const void* p) {
    // LDS aperture: generic addr low 32 bits == LDS byte offset
    return (unsigned)(unsigned long long)(uintptr_t)p;
}
// copy 32 bytes (16 bf16) per lane: two b128 async transfers (INST_OFFSET is
// added to BOTH the LDS and global addresses per ISA 08_async_tensor §4.4)
__device__ __forceinline__ void asyncCopy32(unsigned lds, const void* g) {
    unsigned long long ga = (unsigned long long)(uintptr_t)g;
    asm volatile("global_load_async_to_lds_b128 %0, %1, off"
                 :: "v"(lds), "v"(ga) : "memory");
    asm volatile("global_load_async_to_lds_b128 %0, %1, off offset:16"
                 :: "v"(lds), "v"(ga) : "memory");
}
__device__ __forceinline__ void waitAsync0() {
    asm volatile("s_wait_asynccnt 0x0" ::: "memory");
}

#define EMB    1024
#define HEADS  8
#define HD     128
#define SEQ    2048
#define BATCH  4
#define ROWS   (BATCH * SEQ)  // 8192 tokens
#define FFDIM  (4 * EMB)      // 4096
#define LNEPS  1e-5f

// ---------------- fp32 -> bf16 (row-major) ----------------
__global__ void cvt_f32_to_bf16(const float* __restrict__ in, __bf16* __restrict__ out, int n) {
    for (int i = blockIdx.x * blockDim.x + threadIdx.x; i < n; i += gridDim.x * blockDim.x)
        out[i] = (__bf16)in[i];
}

// ------------- fp32 [K][N] -> bf16 transposed [N][K] (LDS-tiled) -------------
__global__ __launch_bounds__(256) void cvt_transpose_bf16(
    const float* __restrict__ in, __bf16* __restrict__ out, int K, int N)
{
    __shared__ __bf16 tile[32][33];
    const int k0 = blockIdx.y * 32, n0 = blockIdx.x * 32;
    const int tr  = threadIdx.x >> 3;       // 0..31
    const int tc4 = (threadIdx.x & 7) * 4;  // 0,4,..,28
    const float4 f = *(const float4*)(in + (size_t)(k0 + tr) * N + n0 + tc4);
    tile[tr][tc4 + 0] = (__bf16)f.x;
    tile[tr][tc4 + 1] = (__bf16)f.y;
    tile[tr][tc4 + 2] = (__bf16)f.z;
    tile[tr][tc4 + 3] = (__bf16)f.w;
    __syncthreads();
    __bf16* q = out + (size_t)(n0 + tr) * K + k0 + tc4;
    q[0] = tile[tc4 + 0][tr];
    q[1] = tile[tc4 + 1][tr];
    q[2] = tile[tc4 + 2][tr];
    q[3] = tile[tc4 + 3][tr];
}

// ---------------- WMMA GEMM: C[M,N] = A[M,K] @ Bt[N,K]^T + bias ----------------
// Block tile 128x128, K-step 32, double-buffered async-to-LDS staging.
// 8 waves: 2 along M (64 rows) x 4 along N (32 cols); wave does 4x2 16x16 tiles.
// MODE 0: f32 out.  MODE 1: GELU -> bf16 out.  MODE 2: QKV de-interleave scatter
//         (q,k row-major [b,h,n,d]; v TRANSPOSED [b,h,d,n]) -> bf16.
template <int MODE>
__global__ __launch_bounds__(256) void gemm_bf16_wmma(
    const __bf16* __restrict__ A,   // [M][K] bf16
    const __bf16* __restrict__ Bt,  // [N][K] bf16 (pre-transposed weights)
    const float* __restrict__ bias,
    float* __restrict__ Cf, __bf16* __restrict__ Cb,
    int M, int N, int K)
{
    __shared__ __bf16 As[2][128][32];   // [m][k]
    __shared__ __bf16 Bs[2][128][32];   // [n][k]

    const int t    = threadIdx.x;
    const int wave = t >> 5;
    const int lane = t & 31;
    const int lh   = lane >> 4;
    const int ln   = lane & 15;
    const int wm   = (wave & 1) * 64;
    const int wn   = (wave >> 1) * 32;
    const int m0   = blockIdx.y * 128;
    const int n0   = blockIdx.x * 128;

    // staging: 2 threads/row, 16 bf16 (32B) each; A and Bt tiles are isomorphic
    const int sr = t >> 1, sc = (t & 1) * 16;
    const __bf16* aRow = A  + (size_t)(m0 + sr) * K + sc;
    const __bf16* bRow = Bt + (size_t)(n0 + sr) * K + sc;

    v8f acc[4][2];
#pragma unroll
    for (int i = 0; i < 4; ++i)
#pragma unroll
        for (int j = 0; j < 2; ++j) acc[i][j] = vzero();

    const int nk = K >> 5;
    // prologue: DMA tile 0
    asyncCopy32(ldsOff(&As[0][sr][sc]), aRow);
    asyncCopy32(ldsOff(&Bs[0][sr][sc]), bRow);

    for (int kt = 0; kt < nk; ++kt) {
        const int buf = kt & 1;
        waitAsync0();          // my tile-kt DMAs have landed in LDS
        __syncthreads();       // everyone's landed; prev compute on buf^1 done
        if (kt + 1 < nk) {     // overlap: DMA tile kt+1 while computing kt
            asyncCopy32(ldsOff(&As[buf ^ 1][sr][sc]), aRow + (kt + 1) * 32);
            asyncCopy32(ldsOff(&Bs[buf ^ 1][sr][sc]), bRow + (kt + 1) * 32);
        }

        U16 af[4];
#pragma unroll
        for (int i = 0; i < 4; ++i) {
            const __bf16* p = &As[buf][wm + i * 16 + ln][lh * 8];
            af[i].h[0] = *(const v8bf*)p;         // K = kb..kb+7
            af[i].h[1] = *(const v8bf*)(p + 16);  // K = kb+16..kb+23
        }
        U16 bf2[2];
#pragma unroll
        for (int j = 0; j < 2; ++j) {
            const __bf16* p = &Bs[buf][wn + j * 16 + ln][lh * 8];
            bf2[j].h[0] = *(const v8bf*)p;        // K = kb..kb+7
            bf2[j].h[1] = *(const v8bf*)(p + 16); // K = kb+16..kb+23
        }
#pragma unroll
        for (int i = 0; i < 4; ++i)
#pragma unroll
            for (int j = 0; j < 2; ++j)
                acc[i][j] = wmma_bf16(af[i].v, bf2[j].v, acc[i][j]);
    }

    // ---- epilogue: C layout row = i*16 + lh*8 + r, col = lane%16 ----
#pragma unroll
    for (int i = 0; i < 4; ++i)
#pragma unroll
        for (int j = 0; j < 2; ++j) {
            const int col = n0 + wn + j * 16 + ln;
            const float bs = bias ? bias[col] : 0.0f;
#pragma unroll
            for (int r = 0; r < 8; ++r) {
                const int row = m0 + wm + i * 16 + lh * 8 + r;
                float vv = acc[i][j][r] + bs;
                if constexpr (MODE == 0) {
                    Cf[(size_t)row * N + col] = vv;
                } else if constexpr (MODE == 1) {
                    float gl = 0.5f * vv * (1.0f + erff(vv * 0.70710678118654752f));
                    Cb[(size_t)row * N + col] = (__bf16)gl;
                } else {
                    // col = h*384 + dd*3 + which
                    const int which = col % 3;
                    const int rem   = col / 3;    // h*128 + dd
                    const int dd    = rem & 127;
                    const int hh    = rem >> 7;
                    const int bb    = row >> 11;
                    const int nn    = row & 2047;
                    size_t idx;
                    if (which == 2)  // V stored transposed [b,h,d,n] for async staging
                        idx = (size_t)2 * ((size_t)ROWS * EMB) +
                              (((size_t)(bb * HEADS + hh) * HD + dd) * SEQ + nn);
                    else
                        idx = (size_t)which * ((size_t)ROWS * EMB) +
                              (((size_t)(bb * HEADS + hh) * SEQ + nn) * HD + dd);
                    Cb[idx] = (__bf16)vv;
                }
            }
        }
}

// ---------------- Flash attention: o = (softmax(QK^T)/sqrt(EMB)) @ V ----------------
// q,k: [b,h,n,d] bf16; vt: [b,h,d,n] bf16 (pre-transposed).
// grid: (SEQ/128, BATCH*HEADS); block 256 (8 waves x 16 q-rows).
__global__ __launch_bounds__(256) void attn_flash(
    const __bf16* __restrict__ q, const __bf16* __restrict__ k,
    const __bf16* __restrict__ vt, __bf16* __restrict__ o)
{
    __shared__ __bf16 Ks[2][32][128];   // [kv][d]
    __shared__ __bf16 Vt[2][128][32];   // [d][kv]
    __shared__ __bf16 Ps[8][16][32];    // per-wave P re-layout scratch

    const int t    = threadIdx.x;
    const int wave = t >> 5;
    const int lane = t & 31;
    const int lh   = lane >> 4;
    const int ln   = lane & 15;
    const int bh   = blockIdx.y;
    const int q0   = blockIdx.x * 128 + wave * 16;

    // resident Q fragments (4 k-steps of 32 across d=128)
    U16 qa[4];
    const __bf16* qp = q + ((size_t)bh * SEQ + q0 + ln) * HD;
#pragma unroll
    for (int s = 0; s < 4; ++s) {
        qa[s].h[0] = *(const v8bf*)(qp + s * 32 + lh * 8);
        qa[s].h[1] = *(const v8bf*)(qp + s * 32 + lh * 8 + 16);
    }

    v8f oacc[8];
#pragma unroll
    for (int dt = 0; dt < 8; ++dt) oacc[dt] = vzero();
    float mrow[8], lrow[8];
#pragma unroll
    for (int r = 0; r < 8; ++r) { mrow[r] = -1e30f; lrow[r] = 0.0f; }

    // staging geometry: K tile 32x128 (8 thr/row, 32B); Vt tile 128x32 (2 thr/row, 32B)
    const int kr = t >> 3, kc = (t & 7) * 16;
    const int vr = t >> 1, vc = (t & 1) * 16;
    const __bf16* kRow = k  + ((size_t)bh * SEQ + kr) * HD + kc;   // + j*32*HD
    const __bf16* vRow = vt + ((size_t)bh * HD + vr) * SEQ + vc;   // + j*32

    const int nj = SEQ / 32;
    asyncCopy32(ldsOff(&Ks[0][kr][kc]), kRow);
    asyncCopy32(ldsOff(&Vt[0][vr][vc]), vRow);

    for (int j = 0; j < nj; ++j) {
        const int buf = j & 1;
        waitAsync0();
        __syncthreads();
        if (j + 1 < nj) {
            asyncCopy32(ldsOff(&Ks[buf ^ 1][kr][kc]), kRow + (size_t)(j + 1) * 32 * HD);
            asyncCopy32(ldsOff(&Vt[buf ^ 1][vr][vc]), vRow + (j + 1) * 32);
        }

        // S = Q @ K^T (B-frag columns are contiguous K rows)
        v8f sacc[2];
        sacc[0] = vzero(); sacc[1] = vzero();
#pragma unroll
        for (int nt = 0; nt < 2; ++nt) {
            const __bf16* kb = &Ks[buf][nt * 16 + ln][0];
#pragma unroll
            for (int s = 0; s < 4; ++s) {
                U16 bfr;
                bfr.h[0] = *(const v8bf*)(kb + s * 32 + lh * 8);
                bfr.h[1] = *(const v8bf*)(kb + s * 32 + lh * 8 + 16);
                sacc[nt] = wmma_bf16(qa[s].v, bfr.v, sacc[nt]);
            }
        }

        // online softmax (row = vgpr r + 8*lh, cols striped over 16 lanes)
#pragma unroll
        for (int r = 0; r < 8; ++r) {
            float a0 = sacc[0][r], a1 = sacc[1][r];
            float mc = fmaxf(a0, a1);
#pragma unroll
            for (int msk = 8; msk >= 1; msk >>= 1) mc = fmaxf(mc, __shfl_xor(mc, msk, 32));
            float mn    = fmaxf(mrow[r], mc);
            float alpha = __expf(mrow[r] - mn);
            float p0 = __expf(a0 - mn), p1 = __expf(a1 - mn);
            float rs = p0 + p1;
#pragma unroll
            for (int msk = 8; msk >= 1; msk >>= 1) rs += __shfl_xor(rs, msk, 32);
            lrow[r] = lrow[r] * alpha + rs;
            mrow[r] = mn;
#pragma unroll
            for (int dt = 0; dt < 8; ++dt) oacc[dt][r] *= alpha;
            Ps[wave][lh * 8 + r][ln]      = (__bf16)p0;
            Ps[wave][lh * 8 + r][16 + ln] = (__bf16)p1;
        }

        // P re-layout (same-wave LDS ops are in order; DScnt waits auto-inserted)
        U16 pf;
        pf.h[0] = *(const v8bf*)&Ps[wave][ln][lh * 8];
        pf.h[1] = *(const v8bf*)&Ps[wave][ln][lh * 8 + 16];
#pragma unroll
        for (int dt = 0; dt < 8; ++dt) {
            U16 vf;
            const __bf16* vb = &Vt[buf][dt * 16 + ln][lh * 8];
            vf.h[0] = *(const v8bf*)vb;
            vf.h[1] = *(const v8bf*)(vb + 16);
            oacc[dt] = wmma_bf16(pf.v, vf.v, oacc[dt]);
        }
    }

    // finalize: 1/l and the reference's post-softmax 1/sqrt(EMB)=1/32 scale
    const int bb = bh >> 3, hh = bh & 7;
#pragma unroll
    for (int r = 0; r < 8; ++r) {
        float inv = 1.0f / (lrow[r] * 32.0f);
        const int qr = q0 + lh * 8 + r;
#pragma unroll
        for (int dt = 0; dt < 8; ++dt) {
            const int col = hh * HD + dt * 16 + ln;
            o[((size_t)bb * SEQ + qr) * EMB + col] = (__bf16)(oacc[dt][r] * inv);
        }
    }
}

// ---------------- fused residual add + LayerNorm (f32 out + optional bf16 out) ----------------
__global__ __launch_bounds__(256) void add_layernorm(
    const float* __restrict__ x, const float* __restrict__ y,
    const float* __restrict__ g, const float* __restrict__ b,
    float* __restrict__ out, __bf16* __restrict__ out_bf)
{
    const int row = blockIdx.x;
    const int t   = threadIdx.x;
    __shared__ float rS[8], rS2[8];

    const float4 xv = *(const float4*)(x + (size_t)row * EMB + t * 4);
    const float4 yv = *(const float4*)(y + (size_t)row * EMB + t * 4);
    float vals[4] = {xv.x + yv.x, xv.y + yv.y, xv.z + yv.z, xv.w + yv.w};
    float s  = vals[0] + vals[1] + vals[2] + vals[3];
    float ss = vals[0] * vals[0] + vals[1] * vals[1] + vals[2] * vals[2] + vals[3] * vals[3];
#pragma unroll
    for (int msk = 16; msk >= 1; msk >>= 1) {
        s  += __shfl_xor(s, msk, 32);
        ss += __shfl_xor(ss, msk, 32);
    }
    if ((t & 31) == 0) { rS[t >> 5] = s; rS2[t >> 5] = ss; }
    __syncthreads();
    float S = 0.f, SS = 0.f;
#pragma unroll
    for (int w = 0; w < 8; ++w) { S += rS[w]; SS += rS2[w]; }
    const float mu   = S * (1.0f / EMB);
    const float var  = SS * (1.0f / EMB) - mu * mu;
    const float rstd = rsqrtf(var + LNEPS);
#pragma unroll
    for (int i = 0; i < 4; ++i) {
        const int c = t * 4 + i;
        const float r = (vals[i] - mu) * rstd * g[c] + b[c];
        out[(size_t)row * EMB + c] = r;
        if (out_bf) out_bf[(size_t)row * EMB + c] = (__bf16)r;
    }
}

// ---------------- orchestration ----------------
extern "C" void kernel_launch(void* const* d_in, const int* in_sizes, int n_in,
                              void* d_out, int out_size, void* d_ws, size_t ws_size,
                              hipStream_t stream)
{
    const float* x      = (const float*)d_in[0];
    const float* wqkv_w = (const float*)d_in[1];
    const float* wqkv_b = (const float*)d_in[2];
    const float* proj_w = (const float*)d_in[3];
    const float* proj_b = (const float*)d_in[4];
    const float* lin1_w = (const float*)d_in[5];
    const float* lin1_b = (const float*)d_in[6];
    const float* lin2_w = (const float*)d_in[7];
    const float* lin2_b = (const float*)d_in[8];
    const float* ln_g   = (const float*)d_in[9];
    const float* ln_b   = (const float*)d_in[10];

    uint8_t* wsp = (uint8_t*)d_ws;
    auto take = [&](size_t bytes) -> void* {
        void* p = (void*)wsp;
        wsp += (bytes + 255) & ~(size_t)255;
        return p;
    };

    __bf16* x_bf    = (__bf16*)take((size_t)ROWS * EMB * 2);
    __bf16* wqkvT   = (__bf16*)take((size_t)EMB * 3 * EMB * 2);  // [3E][E]
    __bf16* projT   = (__bf16*)take((size_t)EMB * EMB * 2);      // [E][E]
    __bf16* lin1T   = (__bf16*)take((size_t)EMB * FFDIM * 2);    // [4E][E]
    __bf16* lin2T   = (__bf16*)take((size_t)FFDIM * EMB * 2);    // [E][4E]
    __bf16* qkv_s   = (__bf16*)take((size_t)3 * ROWS * EMB * 2); // q|k|[v transposed]
    __bf16* o_bf    = (__bf16*)take((size_t)ROWS * EMB * 2);
    float*  mha     = (float*)take((size_t)ROWS * EMB * 4);
    float*  ln1     = (float*)take((size_t)ROWS * EMB * 4);
    __bf16* ln1_bf  = (__bf16*)take((size_t)ROWS * EMB * 2);
    __bf16* ff1     = (__bf16*)take((size_t)ROWS * FFDIM * 2);
    float*  ff2     = (float*)take((size_t)ROWS * EMB * 4);

    cvt_f32_to_bf16<<<512, 256, 0, stream>>>(x, x_bf, ROWS * EMB);
    cvt_transpose_bf16<<<dim3(3 * EMB / 32, EMB / 32), 256, 0, stream>>>(wqkv_w, wqkvT, EMB, 3 * EMB);
    cvt_transpose_bf16<<<dim3(EMB / 32, EMB / 32), 256, 0, stream>>>(proj_w, projT, EMB, EMB);
    cvt_transpose_bf16<<<dim3(FFDIM / 32, EMB / 32), 256, 0, stream>>>(lin1_w, lin1T, EMB, FFDIM);
    cvt_transpose_bf16<<<dim3(EMB / 32, FFDIM / 32), 256, 0, stream>>>(lin2_w, lin2T, FFDIM, EMB);

    // QKV GEMM with de-interleave scatter epilogue (V written transposed)
    gemm_bf16_wmma<2><<<dim3(3 * EMB / 128, ROWS / 128), 256, 0, stream>>>(
        x_bf, wqkvT, wqkv_b, nullptr, qkv_s, ROWS, 3 * EMB, EMB);

    attn_flash<<<dim3(SEQ / 128, BATCH * HEADS), 256, 0, stream>>>(
        qkv_s, qkv_s + (size_t)ROWS * EMB, qkv_s + (size_t)2 * ROWS * EMB, o_bf);

    gemm_bf16_wmma<0><<<dim3(EMB / 128, ROWS / 128), 256, 0, stream>>>(
        o_bf, projT, proj_b, mha, nullptr, ROWS, EMB, EMB);

    add_layernorm<<<ROWS, 256, 0, stream>>>(x, mha, ln_g, ln_b, ln1, ln1_bf);

    gemm_bf16_wmma<1><<<dim3(FFDIM / 128, ROWS / 128), 256, 0, stream>>>(
        ln1_bf, lin1T, lin1_b, nullptr, ff1, ROWS, FFDIM, EMB);
    gemm_bf16_wmma<0><<<dim3(EMB / 128, ROWS / 128), 256, 0, stream>>>(
        ff1, lin2T, lin2_b, ff2, nullptr, ROWS, EMB, FFDIM);

    add_layernorm<<<ROWS, 256, 0, stream>>>(ln1, ff2, ln_g, ln_b, (float*)d_out, nullptr);
}
